// ImportanceRenderer_39565238731036
// MI455X (gfx1250) — matrix operand order repaired and checked by reading, not compiled, not run on hardware
//
#include <hip/hip_runtime.h>
#include <math.h>

typedef __attribute__((ext_vector_type(16))) _Float16 v16h;
typedef __attribute__((ext_vector_type(8)))  _Float16 v8h;
typedef __attribute__((ext_vector_type(8)))  float    v8f;

namespace {
constexpr int NPL   = 32;    // image planes
constexpr int SD    = 48;    // depth_resolution == N_importance
constexpr int HW    = 64;    // plane H == W
constexpr int FDIM  = 160;   // 96 feats + 64 pix coords
constexpr int HID   = 64;
constexpr int ODIM  = 33;    // 1 sigma + 32 rgb
constexpr int OPAD  = 48;    // pad N to 3 WMMA tiles
constexpr int PTS   = 64;    // points per block
constexpr int NTHR  = 64;    // 2 waves
constexpr int NWAVES = 2;

__device__ __forceinline__ float softplusf(float x) {
  return fmaxf(x, 0.f) + log1pf(expf(-fabsf(x)));
}

// Build a v16h A/B fragment from two contiguous 16-byte LDS runs.
// A layout (ISA 7.12.2, 16-bit A 16x32): lane holds K = 32kt+8h..+7 then +16..+23.
// B layout (transposed weights in LDS): lane holds K = 32kt+8h..+7 then +16..+23 (same as A).
__device__ __forceinline__ v16h frag2(const _Float16* p0, const _Float16* p1) {
  v8h lo = *(const v8h*)p0;
  v8h hi = *(const v8h*)p1;
  return __builtin_shufflevector(lo, hi, 0, 1, 2, 3, 4, 5, 6, 7,
                                 8, 9, 10, 11, 12, 13, 14, 15);
}
__device__ __forceinline__ v16h fragc(const _Float16* p) { return frag2(p, p + 8); }

__device__ void mm4(const double* A, const double* B, double* C) {
  for (int i = 0; i < 4; ++i)
    for (int j = 0; j < 4; ++j) {
      double s = 0.0;
      for (int k = 0; k < 4; ++k) s += A[i * 4 + k] * B[k * 4 + j];
      C[i * 4 + j] = s;
    }
}
} // namespace

// ---------------- K0: camera poses (fp64, faithful to numpy) + min/max init ----
__global__ void pose_kernel(float* __restrict__ poses, unsigned* __restrict__ minmax) {
  int i = threadIdx.x;
  if (i == 0) { minmax[0] = 0x7F800000u; minmax[1] = 0u; }
  if (i >= NPL) return;
  const double PI = 3.14159265358979323846;
  double y = 1.0 - (i / (double)(NPL - 1)) * 2.0;
  double rr = sqrt(fmax(0.0, 1.0 - y * y));
  double golden = PI * (sqrt(5.0) - 1.0);
  double th = golden * (double)i;
  double x = cos(th) * rr, z = sin(th) * rr;
  double phi = atan2(z, sqrt(x * x + y * y));
  double theta = atan2(y, x);
  double radius = -1.307;
  double T[16]  = {1,0,0,0, 0,1,0,0, 0,0,1,radius, 0,0,0,1};
  double p = phi / 180.0 * PI;
  double RP[16] = {1,0,0,0, 0,cos(p),-sin(p),0, 0,sin(p),cos(p),0, 0,0,0,1};
  double t = theta / 180.0 * PI;
  double RT[16] = {cos(t),0,-sin(t),0, 0,1,0,0, sin(t),0,cos(t),0, 0,0,0,1};
  double e = 90.0 / 180.0 * PI;
  double RE[16] = {cos(e),sin(e),0,0, -sin(e),cos(e),0,0, 0,0,1,0, 0,0,0,1};
  double PRE[16] = {-1,0,0,0, 0,0,1,0, 0,1,0,0, 0,0,0,1};
  double t1[16], t2[16], t3[16], M[16];
  mm4(RP, T, t1); mm4(RT, t1, t2); mm4(RE, t2, t3); mm4(PRE, t3, M);
  // M = pose @ diag(-1,1,1,1): negate column 0
  M[0] = -M[0]; M[4] = -M[4]; M[8] = -M[8]; M[12] = -M[12];
  // affine with orthogonal 3x3 -> inverse = [R^T | -R^T t]
  double R00=M[0],R01=M[1],R02=M[2],tx=M[3];
  double R10=M[4],R11=M[5],R12=M[6],ty=M[7];
  double R20=M[8],R21=M[9],R22=M[10],tz=M[11];
  float* P4 = poses + i * 12;
  P4[0]=(float)R00; P4[1]=(float)R10; P4[2]=(float)R20; P4[3]=(float)(-(R00*tx+R10*ty+R20*tz));
  P4[4]=(float)R01; P4[5]=(float)R11; P4[6]=(float)R21; P4[7]=(float)(-(R01*tx+R11*ty+R21*tz));
  P4[8]=(float)R02; P4[9]=(float)R12; P4[10]=(float)R22; P4[11]=(float)(-(R02*tx+R12*ty+R22*tz));
}

// ---------------- K1: stratified coarse depths ---------------------------------
__global__ void depth_kernel(const float* __restrict__ noise, float* __restrict__ dc, int P) {
  int i = blockIdx.x * blockDim.x + threadIdx.x;
  if (i >= P) return;
  int s = i % SD;
  const float delta = 2.0f / (float)(SD - 1);
  dc[i] = 0.5f + (float)s * delta + noise[i] * delta;
}

// ---------------- K2: fused features + WMMA MLP --------------------------------
__global__ __launch_bounds__(NTHR)
void pass_kernel(const float* __restrict__ planes,
                 const float* __restrict__ rayo,
                 const float* __restrict__ rayd,
                 const float* __restrict__ depths,   // [R*SD]
                 const float* __restrict__ W1, const float* __restrict__ b1,
                 const float* __restrict__ W2, const float* __restrict__ b2,
                 const float* __restrict__ posesG,
                 float* __restrict__ colorsOut,      // [P*32]
                 float* __restrict__ densOut,        // [P]
                 int Nper, int Ptot) {
  __shared__ __align__(16) _Float16 Xs[PTS][FDIM + 8];   // 336B rows (21*16)
  __shared__ __align__(16) _Float16 W1t[HID][FDIM];      // transposed: [col][k], 320B rows
  __shared__ __align__(16) _Float16 W2t[OPAD][HID];      // transposed: [col][k], 128B rows
  __shared__ __align__(16) _Float16 Hs[PTS][HID + 8];    // 144B rows (9*16)
  __shared__ float PoseS[NPL * 12];

  int t = threadIdx.x;
  for (int i = t; i < FDIM * HID; i += NTHR) {
    int r = i / HID, c = i % HID;
    W1t[c][r] = (_Float16)W1[i];
  }
  for (int i = t; i < HID * OPAD; i += NTHR) {
    int r = i / OPAD, c = i % OPAD;
    W2t[c][r] = (_Float16)((c < ODIM) ? W2[r * ODIM + c] : 0.f);
  }
  for (int i = t; i < NPL * 12; i += NTHR) PoseS[i] = posesG[i];
  __syncthreads();

  // ---- feature generation: one point per thread -------------------------------
  int pid  = blockIdx.x * PTS + t;
  int pidc = (pid < Ptot) ? pid : (Ptot - 1);
  int ray  = pidc / SD;
  int b    = ray / Nper;
  float dep = depths[pidc];
  float ox = rayo[ray*3+0], oy = rayo[ray*3+1], oz = rayo[ray*3+2];
  float dx = rayd[ray*3+0], dy = rayd[ray*3+1], dz = rayd[ray*3+2];
  float X = ox + dep * dx, Y = oy + dep * dy, Z = oz + dep * dz;
  const float* pb = planes + (size_t)b * 96 * HW * HW;
  #pragma unroll 1
  for (int k = 0; k < NPL; ++k) {
    const float* kb = pb + (size_t)k * 3 * HW * HW;
    // prefetch next plane's 3-channel block (planes are L2-resident)
    int kn = (k + 1 < NPL) ? (k + 1) : k;
    __builtin_prefetch(pb + (size_t)kn * 3 * HW * HW, 0, 0);
    const float* M = &PoseS[k * 12];
    float cx = M[0]*X + M[1]*Y + M[2] *Z + M[3];
    float cy = M[4]*X + M[5]*Y + M[6] *Z + M[7];
    float cz = M[8]*X + M[9]*Y + M[10]*Z + M[11];
    // KMAT = [[1.0254,0,0.5],[0,1.0254,0.5],[0,0,1]]
    float psx = 1.0254f * cx + 0.5f * cz;
    float psy = 1.0254f * cy + 0.5f * cz;
    float gx = fminf(fmaxf(psx / cz, 0.f), 1.f) * 2.f - 1.f;
    float gy = fminf(fmaxf(psy / cz, 0.f), 1.f) * 2.f - 1.f;
    // grid_sample bilinear, zeros, align_corners=False
    float ix = ((gx + 1.f) * (float)HW - 1.f) * 0.5f;
    float iy = ((gy + 1.f) * (float)HW - 1.f) * 0.5f;
    float x0f = floorf(ix), y0f = floorf(iy);
    float wx = ix - x0f, wy = iy - y0f;
    float f0 = 0.f, f1 = 0.f, f2 = 0.f;
    #pragma unroll
    for (int tap = 0; tap < 4; ++tap) {
      float xx = x0f + (float)(tap & 1);
      float yy = y0f + (float)(tap >> 1);
      float w  = ((tap & 1) ? wx : 1.f - wx) * ((tap >> 1) ? wy : 1.f - wy);
      bool valid = (xx >= 0.f) && (xx <= (float)(HW - 1)) &&
                   (yy >= 0.f) && (yy <= (float)(HW - 1));
      int xi = (int)fminf(fmaxf(xx, 0.f), (float)(HW - 1));
      int yi = (int)fminf(fmaxf(yy, 0.f), (float)(HW - 1));
      float wv = valid ? w : 0.f;
      int off = yi * HW + xi;
      f0 += kb[off]            * wv;
      f1 += kb[HW*HW   + off]  * wv;
      f2 += kb[2*HW*HW + off]  * wv;
    }
    Xs[t][k*3 + 0] = (_Float16)f0;
    Xs[t][k*3 + 1] = (_Float16)f1;
    Xs[t][k*3 + 2] = (_Float16)f2;
    Xs[t][96 + 2*k]     = (_Float16)gx;
    Xs[t][96 + 2*k + 1] = (_Float16)gy;
  }
  __syncthreads();

  int lane  = threadIdx.x & 31;
  int halfi = lane >> 4;
  int wvid  = threadIdx.x >> 5;
  int lcol  = lane & 15;

  // ---- GEMM1: [64,160] x [160,64] + b1, softplus -> Hs ------------------------
  // each wave owns 2 M-tiles; A fragments cached in regs, reused over 4 N-tiles
  #pragma unroll
  for (int mtq = 0; mtq < 2; ++mtq) {
    int mt = wvid * 2 + mtq;
    int arow = mt * 16 + lcol;
    v16h afr[FDIM / 32];
    #pragma unroll
    for (int kt = 0; kt < FDIM / 32; ++kt)
      afr[kt] = frag2(&Xs[arow][kt * 32 + 8 * halfi],
                      &Xs[arow][kt * 32 + 8 * halfi + 16]);
    #pragma unroll
    for (int nt = 0; nt < 4; ++nt) {
      int bcol = nt * 16 + lcol;
      v8f acc = {0.f, 0.f, 0.f, 0.f, 0.f, 0.f, 0.f, 0.f};
      #pragma unroll
      for (int kt = 0; kt < FDIM / 32; ++kt) {
        v16h bf = frag2(&W1t[bcol][kt * 32 + 8 * halfi], &W1t[bcol][kt * 32 + 8 * halfi + 16]);
        acc = __builtin_amdgcn_wmma_f32_16x16x32_f16(false, afr[kt], false, bf,
                                                     (short)0, acc, false, false);
      }
      float bias = b1[bcol];
      #pragma unroll
      for (int r = 0; r < 8; ++r) {
        int row = mt * 16 + r + halfi * 8;
        Hs[row][bcol] = (_Float16)softplusf(acc[r] + bias);
      }
    }
  }
  __syncthreads();

  // ---- GEMM2: [64,64] x [64,48] + b2 -> sigma / rgb ---------------------------
  #pragma unroll
  for (int mtq = 0; mtq < 2; ++mtq) {
    int mt = wvid * 2 + mtq;
    int arow = mt * 16 + lcol;
    v16h afr[HID / 32];
    #pragma unroll
    for (int kt = 0; kt < HID / 32; ++kt)
      afr[kt] = frag2(&Hs[arow][kt * 32 + 8 * halfi],
                      &Hs[arow][kt * 32 + 8 * halfi + 16]);
    #pragma unroll
    for (int nt = 0; nt < 3; ++nt) {
      int bcol = nt * 16 + lcol;
      v8f acc = {0.f, 0.f, 0.f, 0.f, 0.f, 0.f, 0.f, 0.f};
      #pragma unroll
      for (int kt = 0; kt < HID / 32; ++kt) {
        v16h bf = frag2(&W2t[bcol][kt * 32 + 8 * halfi], &W2t[bcol][kt * 32 + 8 * halfi + 16]);
        acc = __builtin_amdgcn_wmma_f32_16x16x32_f16(false, afr[kt], false, bf,
                                                     (short)0, acc, false, false);
      }
      float bias = (bcol < ODIM) ? b2[bcol] : 0.f;
      #pragma unroll
      for (int r = 0; r < 8; ++r) {
        int row = mt * 16 + r + halfi * 8;
        int p2 = blockIdx.x * PTS + row;
        float ov = acc[r] + bias;
        if (p2 < Ptot) {
          if (bcol == 0) {
            densOut[p2] = ov;
          } else if (bcol < ODIM) {
            float sg = 1.f / (1.f + expf(-ov));
            colorsOut[(size_t)p2 * 32 + (bcol - 1)] = sg * 1.002f - 0.001f;
          }
        }
      }
    }
  }
}

// ---------------- K3: coarse march -> weights ----------------------------------
__global__ void march_kernel(const float* __restrict__ dc, const float* __restrict__ sc,
                             float* __restrict__ wc, int R) {
  int r = blockIdx.x * blockDim.x + threadIdx.x;
  if (r >= R) return;
  int base = r * SD;
  float T = 1.f;
  for (int i = 0; i < SD - 1; ++i) {
    float d0 = dc[base + i], d1 = dc[base + i + 1];
    float dm = 0.5f * (sc[base + i] + sc[base + i + 1]) - 1.f;
    float alpha = 1.f - expf(-softplusf(dm) * (d1 - d0));
    wc[r * (SD - 1) + i] = alpha * T;
    T *= (1.f - alpha + 1e-10f);
  }
}

// ---------------- K4: pool smoothing + inverse-CDF importance sampling ---------
__global__ void imp_kernel(const float* __restrict__ dc, const float* __restrict__ wc,
                           const float* __restrict__ pdfu, float* __restrict__ df,
                           unsigned* __restrict__ minmax, int R) {
  int r = blockIdx.x * blockDim.x + threadIdx.x;
  if (r >= R) return;
  const int M = SD - 3;           // 45 pdf weights
  int base = r * SD;
  float w[SD - 1];
  for (int i = 0; i < SD - 1; ++i) w[i] = wc[r * (SD - 1) + i];
  float m[SD];                    // maxpool k=2 s=1 p=1 (with -inf pads)
  m[0] = w[0]; m[SD - 1] = w[SD - 2];
  for (int i = 1; i < SD - 1; ++i) m[i] = fmaxf(w[i - 1], w[i]);
  float wt[SD - 3];
  float total = 0.f;
  for (int j = 0; j < M; ++j) {   // sm[1:-1] + eps
    wt[j] = 0.5f * (m[j + 1] + m[j + 2]) + 0.01f + 1e-5f;
    total += wt[j];
  }
  float cdf[SD - 2];              // 46 entries, leading 0
  cdf[0] = 0.f;
  for (int j = 0; j < M; ++j) cdf[j + 1] = cdf[j] + wt[j] / total;

  float mn = 3.402823466e38f, mx = -3.402823466e38f;
  for (int s = 0; s < SD; ++s) { float d = dc[base + s]; mn = fminf(mn, d); mx = fmaxf(mx, d); }

  for (int s = 0; s < SD; ++s) {
    float u = pdfu[base + s];
    int idx = 0;                  // searchsorted(cdf, u, 'right')
    for (int j = 0; j < M + 1; ++j) idx += (cdf[j] <= u) ? 1 : 0;
    int below = (idx - 1 > 0) ? idx - 1 : 0;
    int above = (idx < M) ? idx : M;
    float cb = cdf[below], ca = cdf[above];
    float bb = dc[base + below], ba = dc[base + above];
    float denom = (ca - cb < 1e-5f) ? 1.f : (ca - cb);
    float smp = bb + (u - cb) / denom * (ba - bb);
    df[base + s] = smp;
    mn = fminf(mn, smp); mx = fmaxf(mx, smp);
  }
  atomicMin(&minmax[0], __float_as_uint(mn));
  atomicMax(&minmax[1], __float_as_uint(mx));
}

// ---------------- K5: merge-sort 96 samples + final march ----------------------
__global__ void final_kernel(const float* __restrict__ dc, const float* __restrict__ df,
                             const float* __restrict__ sc, const float* __restrict__ sf,
                             const float* __restrict__ cc, const float* __restrict__ cf,
                             const unsigned* __restrict__ minmax,
                             float* __restrict__ out, int R) {
  int r = blockIdx.x * blockDim.x + threadIdx.x;
  if (r >= R) return;
  int base = r * SD;
  float dall[2 * SD];
  int   ord[2 * SD];
  for (int i = 0; i < SD; ++i) { dall[i] = dc[base + i]; dall[SD + i] = df[base + i]; }
  for (int i = 0; i < 2 * SD; ++i) ord[i] = i;
  // stable insertion sort (matches jnp.argsort stable)
  for (int i = 1; i < 2 * SD; ++i) {
    float key = dall[i]; int ko = ord[i];
    int j = i - 1;
    while (j >= 0 && dall[j] > key) { dall[j + 1] = dall[j]; ord[j + 1] = ord[j]; --j; }
    dall[j + 1] = key; ord[j + 1] = ko;
  }
  float rgb[32];
  for (int c = 0; c < 32; ++c) rgb[c] = 0.f;
  float pc[32];
  int o0 = ord[0];
  int s0 = (o0 < SD) ? o0 : o0 - SD;
  const float* cb0 = ((o0 < SD) ? cc : cf) + (size_t)(base + s0) * 32;
  for (int c = 0; c < 32; ++c) pc[c] = cb0[c];
  float sprev = (o0 < SD) ? sc[base + o0] : sf[base + o0 - SD];
  float dprev = dall[0];
  float T = 1.f, wsum = 0.f, dacc = 0.f;
  for (int i = 0; i < 2 * SD - 1; ++i) {
    int o1 = ord[i + 1];
    int s1 = (o1 < SD) ? o1 : o1 - SD;
    float dnext = dall[i + 1];
    float snext = (o1 < SD) ? sc[base + o1] : sf[base + o1 - SD];
    const float* cb = ((o1 < SD) ? cc : cf) + (size_t)(base + s1) * 32;
    float dm = 0.5f * (sprev + snext) - 1.f;
    float alpha = 1.f - expf(-softplusf(dm) * (dnext - dprev));
    float w = alpha * T;
    T *= (1.f - alpha + 1e-10f);
    wsum += w;
    dacc += w * 0.5f * (dprev + dnext);
    for (int c = 0; c < 32; ++c) {
      float cv = cb[c];
      rgb[c] += w * 0.5f * (pc[c] + cv);
      pc[c] = cv;
    }
    dprev = dnext; sprev = snext;
  }
  float depth = dacc / wsum;
  if (!(depth == depth)) depth = __builtin_inff();   // NaN -> inf
  float gmin = __uint_as_float(minmax[0]);
  float gmax = __uint_as_float(minmax[1]);
  depth = fminf(fmaxf(depth, gmin), gmax);
  float* o = out + (size_t)r * 34;
  for (int c = 0; c < 32; ++c) o[c] = rgb[c] * 2.f - 1.f;
  o[32] = depth;
  o[33] = wsum;
}

// ---------------- launch -------------------------------------------------------
extern "C" void kernel_launch(void* const* d_in, const int* in_sizes, int n_in,
                              void* d_out, int out_size, void* d_ws, size_t ws_size,
                              hipStream_t stream) {
  const float* planes = (const float*)d_in[0];
  const float* rayo   = (const float*)d_in[1];
  const float* rayd   = (const float*)d_in[2];
  const float* noise  = (const float*)d_in[3];
  const float* pdfu   = (const float*)d_in[4];
  const float* W1     = (const float*)d_in[5];
  const float* b1     = (const float*)d_in[6];
  const float* W2     = (const float*)d_in[7];
  const float* b2     = (const float*)d_in[8];

  int BN = in_sizes[1] / 3;                 // B*N rays
  int B  = in_sizes[0] / (96 * HW * HW);
  int N  = BN / B;
  int R  = BN;
  int P  = R * SD;

  float*    ws     = (float*)d_ws;
  float*    poses  = ws;                                 // 384 floats
  unsigned* minmax = (unsigned*)(ws + 384);              // 2 uints
  float*    dc     = ws + 512;                           // [P]
  float*    df     = dc + P;                             // [P]
  float*    sc     = df + P;                             // [P]
  float*    sf     = sc + P;                             // [P]
  float*    wc     = sf + P;                             // [R*(SD-1)]
  float*    cc     = wc + (size_t)R * (SD - 1);          // [P*32]
  float*    cf     = cc + (size_t)P * 32;                // [P*32]

  pose_kernel<<<1, 32, 0, stream>>>(poses, minmax);
  depth_kernel<<<(P + 255) / 256, 256, 0, stream>>>(noise, dc, P);
  pass_kernel<<<(P + PTS - 1) / PTS, NTHR, 0, stream>>>(
      planes, rayo, rayd, dc, W1, b1, W2, b2, poses, cc, sc, N, P);
  march_kernel<<<(R + 127) / 128, 128, 0, stream>>>(dc, sc, wc, R);
  imp_kernel<<<(R + 63) / 64, 64, 0, stream>>>(dc, wc, pdfu, df, minmax, R);
  pass_kernel<<<(P + PTS - 1) / PTS, NTHR, 0, stream>>>(
      planes, rayo, rayd, df, W1, b1, W2, b2, poses, cf, sf, N, P);
  final_kernel<<<(R + 63) / 64, 64, 0, stream>>>(
      dc, df, sc, sf, cc, cf, minmax, (float*)d_out, R);
}
